// MultiHeadAttentionConcise_39642548142086
// MI455X (gfx1250) — matrix operand order repaired and checked by reading, not compiled, not run on hardware
//
#include <hip/hip_runtime.h>
#include <math.h>
#include <stdint.h>

// ---------------------------------------------------------------------------
// Problem constants
// ---------------------------------------------------------------------------
#define BB   8
#define LL   1000
#define DD   1024
#define HH   16
#define HD   64
#define LPAD 1024         // padded key length for V^T (zero filled)
#define MROWS (BB*LL)     // 8000, multiple of 64

typedef __attribute__((ext_vector_type(16))) __bf16 bf16x16;
typedef __attribute__((ext_vector_type(8)))  float  f32x8;

union FragU  { bf16x16 v; uint4 q[2]; };
union Half8U { uint4 q; __bf16 e[8]; };

// ---------------------------------------------------------------------------
// gfx1250 async global->LDS copy (ASYNCcnt-tracked), 16B per lane.
// Falls back to VGPR round-trip copy if disabled.
// ---------------------------------------------------------------------------
#define USE_ASYNC_LDS 1

#if USE_ASYNC_LDS
__device__ __forceinline__ void cp16_g2l(__bf16* dst_lds, const __bf16* src_g) {
    // low 32 bits of a generic pointer into LDS == DS byte address
    unsigned ldsoff = (unsigned)(unsigned long long)dst_lds;
    asm volatile("global_load_async_to_lds_b128 %0, %1, off"
                 :: "v"(ldsoff), "v"(src_g) : "memory");
}
__device__ __forceinline__ void wait_async_lds() {
    asm volatile("s_wait_asynccnt 0x0" ::: "memory");
}
#else
__device__ __forceinline__ void cp16_g2l(__bf16* dst_lds, const __bf16* src_g) {
    *(uint4*)dst_lds = *(const uint4*)src_g;
}
__device__ __forceinline__ void wait_async_lds() {}
#endif

// ---------------------------------------------------------------------------
// WMMA wrapper:  D(16x16,f32) = A(16x32,bf16) * B(32x16,bf16) + C
// ---------------------------------------------------------------------------
__device__ __forceinline__ f32x8 wmma_bf16(bf16x16 a, bf16x16 b, f32x8 c) {
    return __builtin_amdgcn_wmma_f32_16x16x32_bf16(
        /*neg_a=*/false, a, /*neg_b=*/false, b,
        /*c_mod=*/(short)0, c, /*reuse_a=*/false, /*reuse_b=*/false);
}

// A-fragment (16x32, bf16): lane m = tid&15, half = (tid>>4)&1.
// elems 0..7  -> K = koff + half*8 + i ; elems 8..15 -> K = koff + 16 + half*8 + i
__device__ __forceinline__ bf16x16 frag_a_ptr(const __bf16* rowp, int koff) {
    const int half = (threadIdx.x >> 4) & 1;
    const __bf16* p = rowp + koff + half * 8;
    FragU f;
    f.q[0] = *(const uint4*)(p);
    f.q[1] = *(const uint4*)(p + 16);
    return f.v;
}

// B-fragment (32x16, bf16) from column-major storage (colp = column base):
// lane n = tid&15 holds column n; elems 0..7 -> K = koff + half*8 + i ; elems 8..15 -> K = koff + 16 + half*8 + i
__device__ __forceinline__ bf16x16 frag_b_ptr(const __bf16* colp, int koff) {
    const int half = (threadIdx.x >> 4) & 1;
    const __bf16* p = colp + koff + half * 8;
    FragU f;
    f.q[0] = *(const uint4*)(p);
    f.q[1] = *(const uint4*)(p + 16);
    return f.v;
}

__device__ __forceinline__ bf16x16 lda_frag(const __bf16* tile, int row0, int ld, int koff) {
    const int m = threadIdx.x & 15;
    return frag_a_ptr(tile + (row0 + m) * ld, koff);
}
__device__ __forceinline__ bf16x16 ldb_frag(const __bf16* tile, int col0, int ld, int koff) {
    const int n = threadIdx.x & 15;
    return frag_b_ptr(tile + (col0 + n) * ld, koff);
}

// ---------------------------------------------------------------------------
// Elementwise helpers
// ---------------------------------------------------------------------------
__global__ void cvt_f32_bf16_x4(const float* __restrict__ in, __bf16* __restrict__ out, int n4) {
    int i = blockIdx.x * blockDim.x + threadIdx.x;
    if (i >= n4) return;
    float4 f = ((const float4*)in)[i];
    Half8U u;  // use first 4
    u.e[0] = (__bf16)f.x; u.e[1] = (__bf16)f.y; u.e[2] = (__bf16)f.z; u.e[3] = (__bf16)f.w;
    ((uint2*)out)[i] = make_uint2(u.q.x, u.q.y);
}

__global__ void zero_u32(uint32_t* __restrict__ p, int n) {
    int i = blockIdx.x * blockDim.x + threadIdx.x;
    if (i < n) p[i] = 0u;
}

// Fused transpose + f32->bf16 convert:  W[K][N] (f32) -> Wt[N][K] (bf16)
// 32x32 tiles via LDS; grid = (N/32, K/32), block = 256.
__global__ __launch_bounds__(256)
void transpose_cvt_kernel(const float* __restrict__ W, __bf16* __restrict__ Wt,
                          int K, int N)
{
    __shared__ __bf16 tileT[32][33];
    const int tx = threadIdx.x & 31;
    const int ty = threadIdx.x >> 5;      // 0..7
    const int n0 = blockIdx.x * 32;
    const int k0 = blockIdx.y * 32;
    #pragma unroll
    for (int i = 0; i < 4; ++i) {
        const int k = k0 + ty + i * 8;
        tileT[tx][ty + i * 8] = (__bf16)W[(size_t)k * N + n0 + tx];
    }
    __syncthreads();
    #pragma unroll
    for (int i = 0; i < 4; ++i) {
        const int n = n0 + ty + i * 8;
        Wt[(size_t)n * K + k0 + tx] = tileT[ty + i * 8][tx];
    }
}

// ---------------------------------------------------------------------------
// GEMM tile mainloop: block = 128 threads = 4 waves; tile 64(M) x 128(N);
// wave w computes a 32x64 sub-tile (2x4 WMMA tiles). K stepped by 32.
// A staged row-major [64][40] in LDS; B (pre-transposed weights, [N][K])
// staged as [128][40] — both via contiguous 16B async copies.
// ---------------------------------------------------------------------------
#define LDK 40

__device__ __forceinline__ void gemm_mainloop(
    const __bf16* __restrict__ A,  int lda,    // [M x K] row major (bf16)
    const __bf16* __restrict__ Wt, int ldwt,   // [N x K] row major (bf16, transposed)
    int m0, int n0, int K,
    __bf16* ldsA, __bf16* ldsB, f32x8 acc[2][4])
{
    const int t    = threadIdx.x;
    const int wave = t >> 5;
    const int wm   = (wave & 1) * 32;
    const int wn   = (wave >> 1) * 64;

    for (int kk = 0; kk < K; kk += 32) {
        // ---- stage A tile 64x32 : 256 chunks of 8 bf16, 2/thread ----
        #pragma unroll
        for (int i = 0; i < 2; ++i) {
            int chunk = t + 128 * i;            // 0..255
            int row   = chunk >> 2;
            int c8    = (chunk & 3) * 8;
            cp16_g2l(&ldsA[row * LDK + c8],
                     &A[(size_t)(m0 + row) * lda + kk + c8]);
        }
        // ---- stage B tile 128(n) x 32(k) : 512 chunks, 4/thread ----
        #pragma unroll
        for (int i = 0; i < 4; ++i) {
            int chunk = t + 128 * i;            // 0..511
            int n     = chunk >> 2;
            int c8    = (chunk & 3) * 8;
            cp16_g2l(&ldsB[n * LDK + c8],
                     &Wt[(size_t)(n0 + n) * ldwt + kk + c8]);
        }
        wait_async_lds();
        __syncthreads();

        const bf16x16 af0 = lda_frag(ldsA, wm,      LDK, 0);
        const bf16x16 af1 = lda_frag(ldsA, wm + 16, LDK, 0);
        #pragma unroll
        for (int nt = 0; nt < 4; ++nt) {
            const bf16x16 bfr = ldb_frag(ldsB, wn + nt * 16, LDK, 0);
            acc[0][nt] = wmma_bf16(af0, bfr, acc[0][nt]);
            acc[1][nt] = wmma_bf16(af1, bfr, acc[1][nt]);
        }
        __syncthreads();
    }
}

// ---------------------------------------------------------------------------
// GEMM 1: qkv = Xb[8000,1024] @ Wqkv + bqkv, scattered to
//   Q[b,h,l,64], K[b,h,l,64] (bf16) and V^T[b,h,64,LPAD] (bf16)
// ---------------------------------------------------------------------------
__global__ __launch_bounds__(128)
void gemm_qkv_kernel(const __bf16* __restrict__ X, const __bf16* __restrict__ Wt,
                     const float* __restrict__ bias,
                     __bf16* __restrict__ Qb, __bf16* __restrict__ Kb,
                     __bf16* __restrict__ Vt)
{
    __shared__ __attribute__((aligned(16))) __bf16 ldsA[64 * LDK];
    __shared__ __attribute__((aligned(16))) __bf16 ldsB[128 * LDK];

    const int m0 = blockIdx.y * 64;
    const int n0 = blockIdx.x * 128;
    const int wave = threadIdx.x >> 5;
    const int wm = (wave & 1) * 32, wn = (wave >> 1) * 64;
    const int half = (threadIdx.x >> 4) & 1;
    const int n16  = threadIdx.x & 15;

    f32x8 acc[2][4];
    #pragma unroll
    for (int i = 0; i < 2; ++i)
        #pragma unroll
        for (int j = 0; j < 4; ++j)
            #pragma unroll
            for (int e = 0; e < 8; ++e) acc[i][j][e] = 0.0f;

    gemm_mainloop(X, DD, Wt, DD, m0, n0, DD, ldsA, ldsB, acc);

    #pragma unroll
    for (int mt = 0; mt < 2; ++mt) {
        #pragma unroll
        for (int nt = 0; nt < 4; ++nt) {
            const int gn = n0 + wn + nt * 16 + n16;
            const float bv = bias[gn];
            #pragma unroll
            for (int r = 0; r < 8; ++r) {
                const int gm = m0 + wm + mt * 16 + half * 8 + r;
                const float v = acc[mt][nt][r] + bv;
                const __bf16 vb = (__bf16)v;
                const int b = gm / LL, l = gm - b * LL;
                if (gn < DD) {
                    const int h = gn >> 6, hd = gn & 63;
                    Qb[((size_t)(b * HH + h) * LL + l) * HD + hd] = vb;
                } else if (gn < 2 * DD) {
                    const int c = gn - DD, h = c >> 6, hd = c & 63;
                    Kb[((size_t)(b * HH + h) * LL + l) * HD + hd] = vb;
                } else {
                    const int c = gn - 2 * DD, h = c >> 6, hd = c & 63;
                    Vt[((size_t)(b * HH + h) * HD + hd) * LPAD + l] = vb;
                }
            }
        }
    }
}

// ---------------------------------------------------------------------------
// GEMM 2 (output proj): out = Ab[8000,1024] @ Wo + bo  (f32 out)
// ---------------------------------------------------------------------------
__global__ __launch_bounds__(128)
void gemm_out_kernel(const __bf16* __restrict__ A, const __bf16* __restrict__ Wt,
                     const float* __restrict__ bias, float* __restrict__ out)
{
    __shared__ __attribute__((aligned(16))) __bf16 ldsA[64 * LDK];
    __shared__ __attribute__((aligned(16))) __bf16 ldsB[128 * LDK];

    const int m0 = blockIdx.y * 64;
    const int n0 = blockIdx.x * 128;
    const int wave = threadIdx.x >> 5;
    const int wm = (wave & 1) * 32, wn = (wave >> 1) * 64;
    const int half = (threadIdx.x >> 4) & 1;
    const int n16  = threadIdx.x & 15;

    f32x8 acc[2][4];
    #pragma unroll
    for (int i = 0; i < 2; ++i)
        #pragma unroll
        for (int j = 0; j < 4; ++j)
            #pragma unroll
            for (int e = 0; e < 8; ++e) acc[i][j][e] = 0.0f;

    gemm_mainloop(A, DD, Wt, DD, m0, n0, DD, ldsA, ldsB, acc);

    #pragma unroll
    for (int mt = 0; mt < 2; ++mt) {
        #pragma unroll
        for (int nt = 0; nt < 4; ++nt) {
            const int gn = n0 + wn + nt * 16 + n16;
            const float bv = bias[gn];
            #pragma unroll
            for (int r = 0; r < 8; ++r) {
                const int gm = m0 + wm + mt * 16 + half * 8 + r;
                out[(size_t)gm * DD + gn] = acc[mt][nt][r] + bv;
            }
        }
    }
}

// ---------------------------------------------------------------------------
// Flash attention: one wave per (b, h, 16-row q tile). Causal, online softmax.
//   S = Q K^T * 0.125 ; O = softmax(S) V ; writes bf16 att[8000,1024]
// ---------------------------------------------------------------------------
#define MASKV (-3.0e38f)

__global__ __launch_bounds__(32)
void attn_kernel(const __bf16* __restrict__ Qb, const __bf16* __restrict__ Kb,
                 const __bf16* __restrict__ Vt, __bf16* __restrict__ att)
{
    __shared__ __attribute__((aligned(16))) __bf16 pT[16 * LDK];

    const int lane = threadIdx.x & 31;
    const int half = lane >> 4;
    const int l16  = lane & 15;
    const int tile = blockIdx.x;           // 0..62
    const int h    = blockIdx.y;
    const int b    = blockIdx.z;
    const int r0   = tile * 16;

    const __bf16* Qp = Qb + (size_t)(b * HH + h) * LL * HD;
    const __bf16* Kp = Kb + (size_t)(b * HH + h) * LL * HD;
    const __bf16* Vp = Vt + (size_t)(b * HH + h) * HD * LPAD;

    // Q a-fragments (K = head dim, 64 -> two frags), rows clamped to L-1
    int qrow = r0 + l16; if (qrow > LL - 1) qrow = LL - 1;
    const bf16x16 qa0 = frag_a_ptr(Qp + (size_t)qrow * HD, 0);
    const bf16x16 qa1 = frag_a_ptr(Qp + (size_t)qrow * HD, 32);

    f32x8 o[4];
    float mrow[8], lrow[8];
    #pragma unroll
    for (int n = 0; n < 4; ++n)
        #pragma unroll
        for (int e = 0; e < 8; ++e) o[n][e] = 0.0f;
    #pragma unroll
    for (int r = 0; r < 8; ++r) { mrow[r] = MASKV; lrow[r] = 0.0f; }

    const float scale = 0.125f;               // 1/sqrt(64)
    const int nChunks = ((r0 + 15) >> 5) + 1; // keys 0 .. r0+15, chunks of 32

    for (int c = 0; c < nChunks; ++c) {
        const int k0 = c * 32;
        f32x8 s[2];

        #pragma unroll
        for (int t = 0; t < 2; ++t) {
            int krow = k0 + t * 16 + l16;
            int krc  = krow > LL - 1 ? LL - 1 : krow;
            const bf16x16 kb0 = frag_b_ptr(Kp + (size_t)krc * HD, 0);
            const bf16x16 kb1 = frag_b_ptr(Kp + (size_t)krc * HD, 32);
            #pragma unroll
            for (int e = 0; e < 8; ++e) s[t][e] = 0.0f;
            s[t] = wmma_bf16(qa0, kb0, s[t]);
            s[t] = wmma_bf16(qa1, kb1, s[t]);
            // scale + causal / length mask
            #pragma unroll
            for (int r = 0; r < 8; ++r) {
                const int row = r0 + half * 8 + r;
                const int key = k0 + t * 16 + l16;
                float v = s[t][r] * scale;
                if (key > row || key >= LL) v = MASKV;
                s[t][r] = v;
            }
        }

        // online softmax over the 32-key chunk (row spans 16 lanes of a half)
        #pragma unroll
        for (int r = 0; r < 8; ++r) {
            float mx = fmaxf(s[0][r], s[1][r]);
            mx = fmaxf(mx, __shfl_xor(mx, 1));
            mx = fmaxf(mx, __shfl_xor(mx, 2));
            mx = fmaxf(mx, __shfl_xor(mx, 4));
            mx = fmaxf(mx, __shfl_xor(mx, 8));
            const float mnew = fmaxf(mrow[r], mx);
            const float corr = __expf(mrow[r] - mnew);
            const float p0 = __expf(s[0][r] - mnew);
            const float p1 = __expf(s[1][r] - mnew);
            float ps = p0 + p1;
            ps += __shfl_xor(ps, 1);
            ps += __shfl_xor(ps, 2);
            ps += __shfl_xor(ps, 4);
            ps += __shfl_xor(ps, 8);
            lrow[r] = lrow[r] * corr + ps;
            mrow[r] = mnew;
            #pragma unroll
            for (int n = 0; n < 4; ++n) o[n][r] *= corr;
            s[0][r] = p0;
            s[1][r] = p1;
        }

        // transpose P (16x32) through LDS into an A-fragment
        #pragma unroll
        for (int r = 0; r < 8; ++r) {
            pT[(half * 8 + r) * LDK + l16]      = (__bf16)s[0][r];
            pT[(half * 8 + r) * LDK + 16 + l16] = (__bf16)s[1][r];
        }
        __syncthreads();
        const bf16x16 pa = lda_frag(pT, 0, LDK, 0);
        // O += P @ V  (V^T is column-major-for-B: contiguous keys per lane)
        #pragma unroll
        for (int n = 0; n < 4; ++n) {
            const bf16x16 vb = frag_b_ptr(Vp + (size_t)(n * 16 + l16) * LPAD, k0);
            o[n] = wmma_bf16(pa, vb, o[n]);
        }
        __syncthreads();
    }

    // normalize + store (bf16, [8000, 1024] layout, cols h*64 + hd)
    #pragma unroll
    for (int r = 0; r < 8; ++r) {
        const int l = r0 + half * 8 + r;
        if (l < LL) {
            const float inv = 1.0f / lrow[r];
            const size_t gm = (size_t)b * LL + l;
            #pragma unroll
            for (int n = 0; n < 4; ++n)
                att[gm * DD + h * HD + n * 16 + l16] = (__bf16)(o[n][r] * inv);
        }
    }
}

// ---------------------------------------------------------------------------
// Launch
// ---------------------------------------------------------------------------
extern "C" void kernel_launch(void* const* d_in, const int* in_sizes, int n_in,
                              void* d_out, int out_size, void* d_ws, size_t ws_size,
                              hipStream_t stream) {
    const float* x    = (const float*)d_in[0];   // [8,1000,1024]
    const float* Wqkv = (const float*)d_in[1];   // [1024,3072]
    const float* bqkv = (const float*)d_in[2];   // [3072]
    const float* Wo   = (const float*)d_in[3];   // [1024,1024]
    const float* bo   = (const float*)d_in[4];   // [1024]
    float* out = (float*)d_out;                  // [8,1000,1024] f32

    char* ws = (char*)d_ws;
    const size_t nX  = (size_t)MROWS * DD;            // 8,192,000
    const size_t nWq = (size_t)DD * 3 * DD;           // 3,145,728
    const size_t nWo = (size_t)DD * DD;               // 1,048,576
    const size_t nQ  = (size_t)BB * HH * LL * HD;     // 8,192,000
    const size_t nVt = (size_t)BB * HH * HD * LPAD;   // 8,388,608

    __bf16* xb     = (__bf16*)ws;             ws += nX  * 2;
    __bf16* wqkvT  = (__bf16*)ws;             ws += nWq * 2;   // [3072][1024]
    __bf16* woT    = (__bf16*)ws;             ws += nWo * 2;   // [1024][1024]
    __bf16* Qb     = (__bf16*)ws;             ws += nQ  * 2;
    __bf16* Kb     = (__bf16*)ws;             ws += nQ  * 2;
    __bf16* Vt     = (__bf16*)ws;             ws += nVt * 2;
    __bf16* attb   = (__bf16*)ws;             ws += nX  * 2;

    // 1) f32 -> bf16 conversions (x) and fused transpose-converts (weights)
    cvt_f32_bf16_x4<<<(int)((nX / 4 + 255) / 256), 256, 0, stream>>>(x, xb, (int)(nX / 4));
    transpose_cvt_kernel<<<dim3(3 * DD / 32, DD / 32), 256, 0, stream>>>(Wqkv, wqkvT, DD, 3 * DD);
    transpose_cvt_kernel<<<dim3(DD / 32, DD / 32),     256, 0, stream>>>(Wo,   woT,   DD, DD);

    // 2) zero V^T (padding region must be clean every call)
    zero_u32<<<(int)((nVt / 2 + 255) / 256), 256, 0, stream>>>((uint32_t*)Vt, (int)(nVt / 2));

    // 3) QKV projection -> Q, K, V^T
    gemm_qkv_kernel<<<dim3(3 * DD / 128, MROWS / 64), 128, 0, stream>>>(
        xb, wqkvT, bqkv, Qb, Kb, Vt);

    // 4) causal flash attention -> attb (bf16 [8000,1024])
    attn_kernel<<<dim3((LL + 15) / 16, HH, BB), 32, 0, stream>>>(Qb, Kb, Vt, attb);

    // 5) output projection -> out (f32)
    gemm_out_kernel<<<dim3(DD / 128, MROWS / 64), 128, 0, stream>>>(
        attb, woT, bo, out);
}
